// NODE_23295902614125
// MI455X (gfx1250) — matrix operand order repaired and checked by reading, not compiled, not run on hardware
//
#include <hip/hip_runtime.h>
#include <hip/hip_bf16.h>
#include <stdint.h>

// Problem constants (from reference)
#define B_   512
#define S_   256
#define D_   64
#define H_   256
#define T_   50
#define IN_  129
#define INP_ 160   // IN padded to a multiple of 32 for WMMA K-chunks

typedef __attribute__((ext_vector_type(16))) __bf16 v16bf;
typedef __attribute__((ext_vector_type(8)))  float  v8f;

union ABu { v16bf v; uint32_t u[8]; };

static __device__ __forceinline__ uint16_t f2bf(float f) {
  union { float f; uint32_t u; } a; a.f = f;
  uint32_t r = a.u + 0x7FFFu + ((a.u >> 16) & 1u);   // round-to-nearest-even
  return (uint16_t)(r >> 16);
}
static __device__ __forceinline__ uint32_t pk2bf(float f0, float f1) {
  return (uint32_t)f2bf(f0) | ((uint32_t)f2bf(f1) << 16);
}
static __device__ __forceinline__ v8f wmma_bf16(v16bf a, v16bf b, v8f c) {
  // D(f32 16x16) = A(bf16 16x32) * B(bf16 32x16) + C
  return __builtin_amdgcn_wmma_f32_16x16x32_bf16(false, a, false, b, (short)0, c, false, false);
}
static __device__ __forceinline__ float sigf(float x) {
  return 1.0f / (1.0f + __expf(-x));
}

// Launder a zero through an empty asm so values derived from it cannot be
// hoisted/CSE'd out of the enclosing loop (prevents LICM from hoisting the
// loop-invariant weight-tile loads into registers and spilling to scratch,
// which round 1 showed: scratch_load_b128 "Folded Reload" storms).  Within an
// iteration the compiler still vectorizes/schedules the loads freely.
static __device__ __forceinline__ int opaque_zero() {
  int x = 0;
  asm volatile("" : "+v"(x));
  return x;
}

// Gather a 16x32 bf16 A-tile from LDS (row-major, stride ldk bf16 elems) into
// the A layout matching the swizzled B tiles: lanes 0-15 hold K {0..15}, lanes 16-31 hold
// K {16..31}, two packed bf16 per VGPR.
static __device__ __forceinline__ v16bf loadA(const uint16_t* lds, int ldk, int kbase, int lane) {
  ABu a;
  const int hf = lane >> 4;
  const int m  = lane & 15;
  #pragma unroll
  for (int v = 0; v < 8; ++v) {
    int k = kbase + (hf << 4) + (v << 1);
    a.u[v] = *(const uint32_t*)(lds + m * ldk + k);
  }
  return a.v;
}

// Pre-swizzled B tile: 256 dwords; lane reads its 8 contiguous dwords
// (-> two coalesced b128 loads from global, or wide ds loads from LDS).
// `lz` is the laundered zero that keeps these loads inside the loop.
static __device__ __forceinline__ v16bf loadB(const uint32_t* tiles, int tileIdx, int lane, int lz) {
  ABu b;
  const uint32_t* p = tiles + tileIdx * 256 + lane * 8 + lz;
  #pragma unroll
  for (int v = 0; v < 8; ++v) b.u[v] = p[v];
  return b.v;
}

// ---------------------------------------------------------------------------
// Prep: convert W (N x Kreal f32, row-major) into WMMA-ready bf16 B tiles.
// Tile (nt, kc): element (k, n) with n = nt*16 + (lane&15),
// k = kc*32 + (lane>>4)*16 + 2d (+1).  Zero-padded past Kreal.
// grid = (N/16)*Kchunks blocks, 32 threads.
// ---------------------------------------------------------------------------
__global__ void swizzle_kernel(const float* __restrict__ src, uint32_t* __restrict__ dst,
                               int Kreal, int Kchunks) {
  const int tile = blockIdx.x;
  const int lane = threadIdx.x;
  const int nt = tile / Kchunks;
  const int kc = tile - nt * Kchunks;
  const int n  = nt * 16 + (lane & 15);
  const int kb = kc * 32 + (lane >> 4) * 16;
  #pragma unroll
  for (int d = 0; d < 8; ++d) {
    int k = kb + 2 * d;
    float f0 = (k     < Kreal) ? src[(size_t)n * Kreal + k]     : 0.f;
    float f1 = (k + 1 < Kreal) ? src[(size_t)n * Kreal + k + 1] : 0.f;
    dst[(size_t)tile * 256 + lane * 8 + d] = pk2bf(f0, f1);
  }
}

// ---------------------------------------------------------------------------
// Phase 1: LSTM over S=256 steps + masked mean-pool + 2-layer MLP head.
// One workgroup = 16 batch rows, 8 waves. Wave w owns the SAME 32 columns of
// all 4 gates -> gate nonlinearity applied fully in-register (no shared g).
// Weight tiles stream from L2 every step (832 KB/WG working set cannot be
// LDS/register resident; L2 traffic ~7 GB total is cheap at L2 bandwidth).
// ---------------------------------------------------------------------------
__global__ __launch_bounds__(256) void lstm_kernel(
    const float* __restrict__ x, const int* __restrict__ lengths,
    const float* __restrict__ lfeat, const float* __restrict__ lmask,
    const float* __restrict__ b_ih, const float* __restrict__ b_hh,
    const float* __restrict__ bp1, const float* __restrict__ bp2,
    const uint32_t* __restrict__ wih_t, const uint32_t* __restrict__ whh_t,
    const uint32_t* __restrict__ wp1_t, const uint32_t* __restrict__ wp2_t,
    float* __restrict__ predict, float* __restrict__ lossout)
{
  __shared__ __align__(16) uint16_t s_x[16 * INP_];   // x_t tile, bf16, K-padded
  __shared__ __align__(16) uint16_t s_h[16 * H_];     // h_t as bf16 A-tile
  __shared__ __align__(16) float    s_c[16 * H_];
  __shared__ __align__(16) float    s_pool[16 * H_];
  __shared__ int s_len[16];

  const int tid  = threadIdx.x;
  const int w    = tid >> 5;
  const int lane = tid & 31;
  const int hf   = lane >> 4;
  const int nl   = lane & 15;
  const int b0   = blockIdx.x * 16;

  for (int r = tid; r < 16 * H_; r += 256) { s_c[r] = 0.f; s_pool[r] = 0.f; s_h[r] = 0; }
  if (tid < 16) s_len[tid] = lengths[b0 + tid];
  __syncthreads();

  // Gate biases (b_ih + b_hh) for this lane's two column tiles, hoisted.
  float bi0[2], bf0[2], bg0[2], bo0[2];
  #pragma unroll
  for (int j = 0; j < 2; ++j) {
    int ncol = (w * 2 + j) * 16 + nl;                 // 0..255 within a gate
    bi0[j] = b_ih[ncol]          + b_hh[ncol];
    bf0[j] = b_ih[H_ + ncol]     + b_hh[H_ + ncol];
    bg0[j] = b_ih[2 * H_ + ncol] + b_hh[2 * H_ + ncol];
    bo0[j] = b_ih[3 * H_ + ncol] + b_hh[3 * H_ + ncol];
  }

  for (int t = 0; t < S_; ++t) {
    const int lz = opaque_zero();  // pin this step's weight loads inside the loop

    // Stage x[:, t, :] (16 x 129 -> bf16 16 x 160, zero-padded).
    #pragma unroll
    for (int r = 0; r < 10; ++r) {
      int e = tid + r * 256;                          // 2560 = 10*256 elems
      int m = e / INP_;
      int k = e - m * INP_;
      float v = 0.f;
      if (k < IN_) {
        const float* p = x + ((size_t)(b0 + m) * S_ + t) * IN_ + k;
        v = *p;
        if (t + 1 < S_) __builtin_prefetch(p + IN_, 0, 1);  // next timestep
      }
      s_x[m * INP_ + k] = f2bf(v);
    }
    __syncthreads();

    // g = x_t @ W_ih^T + h @ W_hh^T  (this wave: columns [w*32,w*32+32) of
    // each of the 4 gates => 8 accumulator tiles).
    v8f acc[4][2] = {};
    for (int kc = 0; kc < 5; ++kc) {                  // K over padded IN
      v16bf a = loadA(s_x, INP_, kc * 32, lane);
      #pragma unroll
      for (int g = 0; g < 4; ++g)
        #pragma unroll
        for (int j = 0; j < 2; ++j) {
          v16bf b = loadB(wih_t, (g * 16 + w * 2 + j) * 5 + kc, lane, lz);
          acc[g][j] = wmma_bf16(a, b, acc[g][j]);
        }
    }
    for (int kc = 0; kc < 8; ++kc) {                  // K over H
      v16bf a = loadA(s_h, H_, kc * 32, lane);
      #pragma unroll
      for (int g = 0; g < 4; ++g)
        #pragma unroll
        for (int j = 0; j < 2; ++j) {
          v16bf b = loadB(whh_t, (g * 16 + w * 2 + j) * 8 + kc, lane, lz);
          acc[g][j] = wmma_bf16(a, b, acc[g][j]);
        }
    }
    __syncthreads();   // all waves finished reading s_h / s_x

    // In-register LSTM cell (each (m, ncol) owned by exactly one lane).
    #pragma unroll
    for (int j = 0; j < 2; ++j) {
      int ncol = (w * 2 + j) * 16 + nl;
      #pragma unroll
      for (int i = 0; i < 8; ++i) {
        int m = i + hf * 8;
        float gi = acc[0][j][i] + bi0[j];
        float gf = acc[1][j][i] + bf0[j];
        float gg = acc[2][j][i] + bg0[j];
        float go = acc[3][j][i] + bo0[j];
        float c  = sigf(gf) * s_c[m * H_ + ncol] + sigf(gi) * tanhf(gg);
        float h  = sigf(go) * tanhf(c);
        s_c[m * H_ + ncol] = c;
        s_h[m * H_ + ncol] = f2bf(h);
        if (t < s_len[m]) s_pool[m * H_ + ncol] += h;
      }
    }
    __syncthreads();
  }

  // pooled = pool / len, as bf16 A-tile (reuse s_h).
  for (int r = tid; r < 16 * H_; r += 256) {
    int m = r >> 8;
    s_h[r] = f2bf(s_pool[r] / (float)s_len[m]);
  }
  __syncthreads();

  // MLP layer 1: z = relu(pooled @ Wp1^T + bp1)   (N=256 -> 2 tiles/wave)
  uint16_t* zbf = (uint16_t*)s_c;                     // s_c no longer needed
  {
    const int lz = opaque_zero();
    #pragma unroll
    for (int j = 0; j < 2; ++j) {
      v8f acc = {};
      int nt = w * 2 + j;
      for (int kc = 0; kc < 8; ++kc) {
        v16bf a = loadA(s_h, H_, kc * 32, lane);
        v16bf b = loadB(wp1_t, nt * 8 + kc, lane, lz);
        acc = wmma_bf16(a, b, acc);
      }
      int n = nt * 16 + nl;
      float bias = bp1[n];
      #pragma unroll
      for (int i = 0; i < 8; ++i) {
        int m = i + hf * 8;
        float z = acc[i] + bias;
        zbf[m * H_ + n] = f2bf(z > 0.f ? z : 0.f);
      }
    }
  }
  __syncthreads();

  // MLP layer 2: y0 = z @ Wp2^T + bp2   (N=64 -> waves 0..3, wave-uniform branch)
  if (w < 4) {
    const int lz = opaque_zero();
    v8f acc = {};
    for (int kc = 0; kc < 8; ++kc) {
      v16bf a = loadA(zbf, H_, kc * 32, lane);
      v16bf b = loadB(wp2_t, w * 8 + kc, lane, lz);
      acc = wmma_bf16(a, b, acc);
    }
    int n = w * 16 + nl;
    float bias = bp2[n];
    #pragma unroll
    for (int i = 0; i < 8; ++i) {
      int m = i + hf * 8;
      size_t idx = ((size_t)(b0 + m) * T_) * D_ + n;  // t = 0
      float v = acc[i] + bias;
      predict[idx] = v;
      float d = v - lfeat[idx];
      lossout[idx] = d * d * (1.f - lmask[idx]);
    }
  }
}

// ---------------------------------------------------------------------------
// Phase 2: Neural-ODE RK4.  deriv(y) = relu(y@Wd1^T+bd1)@Wd2^T+bd2.
// One workgroup = 16 batch rows, 4 waves. Wd1/Wd2 tiles resident in LDS.
// ---------------------------------------------------------------------------
#define ODE_SMEM 89408

static __device__ __forceinline__ void ode_deriv(
    const uint16_t* s_ybf, const uint32_t* s_wd1, const uint32_t* s_wd2,
    const float* s_bd1, const float* s_bd2, uint16_t* s_zbf, float* s_k,
    int w, int lane)
{
  const int hf = lane >> 4, nl = lane & 15;
  const int lz = opaque_zero();  // keep LDS tile loads inside the RK4 loops
  // layer 1: 16x64 @ 64x256, wave w owns 4 N-tiles
  v8f acc[4] = {};
  for (int kc = 0; kc < 2; ++kc) {
    v16bf a = loadA(s_ybf, D_, kc * 32, lane);
    #pragma unroll
    for (int j = 0; j < 4; ++j) {
      v16bf b = loadB(s_wd1, (w * 4 + j) * 2 + kc, lane, lz);
      acc[j] = wmma_bf16(a, b, acc[j]);
    }
  }
  #pragma unroll
  for (int j = 0; j < 4; ++j) {
    int n = (w * 4 + j) * 16 + nl;
    float bias = s_bd1[n];
    #pragma unroll
    for (int i = 0; i < 8; ++i) {
      int m = i + hf * 8;
      float z = acc[j][i] + bias;
      s_zbf[m * H_ + n] = f2bf(z > 0.f ? z : 0.f);
    }
  }
  __syncthreads();
  // layer 2: 16x256 @ 256x64, wave w owns 1 N-tile
  v8f acc2 = {};
  for (int kc = 0; kc < 8; ++kc) {
    v16bf a = loadA(s_zbf, H_, kc * 32, lane);
    v16bf b = loadB(s_wd2, w * 8 + kc, lane, lz);
    acc2 = wmma_bf16(a, b, acc2);
  }
  int n = w * 16 + nl;
  float bias = s_bd2[n];
  #pragma unroll
  for (int i = 0; i < 8; ++i) {
    int m = i + hf * 8;
    s_k[m * D_ + n] = acc2[i] + bias;
  }
  __syncthreads();
}

__global__ __launch_bounds__(128) void ode_kernel(
    const float* __restrict__ ltime, const float* __restrict__ lfeat,
    const float* __restrict__ lmask,
    const float* __restrict__ bd1, const float* __restrict__ bd2,
    const uint32_t* __restrict__ wd1_t, const uint32_t* __restrict__ wd2_t,
    float* __restrict__ predict, float* __restrict__ lossout)
{
  extern __shared__ __align__(16) unsigned char smem[];
  uint32_t* s_wd1 = (uint32_t*)(smem);                 // 32 KB
  uint32_t* s_wd2 = (uint32_t*)(smem + 32768);         // 32 KB
  float*    s_y   = (float*)   (smem + 65536);         // 16x64
  float*    s_k   = (float*)   (smem + 69632);         // 16x64
  float*    s_acc = (float*)   (smem + 73728);         // 16x64
  float*    s_bd1 = (float*)   (smem + 77824);         // 256
  float*    s_bd2 = (float*)   (smem + 78848);         // 64
  float*    s_dt  = (float*)   (smem + 79104);         // 16
  uint16_t* s_ybf = (uint16_t*)(smem + 79168);         // 16x64 bf16
  uint16_t* s_zbf = (uint16_t*)(smem + 81216);         // 16x256 bf16

  const int tid  = threadIdx.x;
  const int w    = tid >> 5;
  const int lane = tid & 31;
  const int b0   = blockIdx.x * 16;

  // Resident weights + biases into LDS; y0 from predict[:,0,:].
  for (int r = tid; r < 32 * 256; r += 128) { s_wd1[r] = wd1_t[r]; s_wd2[r] = wd2_t[r]; }
  for (int r = tid; r < H_; r += 128) s_bd1[r] = bd1[r];
  if (tid < D_) s_bd2[tid] = bd2[tid];
  for (int r = tid; r < 16 * D_; r += 128) {
    int m = r >> 6, n = r & 63;
    float v = predict[((size_t)(b0 + m) * T_) * D_ + n];
    s_y[r] = v; s_ybf[r] = f2bf(v);
  }
  __syncthreads();

  for (int it = 0; it < T_ - 1; ++it) {
    if (tid < 16) {
      int b = b0 + tid;
      s_dt[tid] = (ltime[b * T_ + it + 1] - ltime[b * T_ + it]) * 0.125f; // /NSUB
    }
    __syncthreads();

    for (int sub = 0; sub < 8; ++sub) {
      // k1
      ode_deriv(s_ybf, s_wd1, s_wd2, s_bd1, s_bd2, s_zbf, s_k, w, lane);
      #pragma unroll
      for (int r = 0; r < 8; ++r) {
        int e = tid + r * 128; int m = e >> 6;
        float k = s_k[e]; s_acc[e] = k;
        s_ybf[e] = f2bf(s_y[e] + 0.5f * s_dt[m] * k);
      }
      __syncthreads();
      // k2
      ode_deriv(s_ybf, s_wd1, s_wd2, s_bd1, s_bd2, s_zbf, s_k, w, lane);
      #pragma unroll
      for (int r = 0; r < 8; ++r) {
        int e = tid + r * 128; int m = e >> 6;
        float k = s_k[e]; s_acc[e] += 2.f * k;
        s_ybf[e] = f2bf(s_y[e] + 0.5f * s_dt[m] * k);
      }
      __syncthreads();
      // k3
      ode_deriv(s_ybf, s_wd1, s_wd2, s_bd1, s_bd2, s_zbf, s_k, w, lane);
      #pragma unroll
      for (int r = 0; r < 8; ++r) {
        int e = tid + r * 128; int m = e >> 6;
        float k = s_k[e]; s_acc[e] += 2.f * k;
        s_ybf[e] = f2bf(s_y[e] + s_dt[m] * k);
      }
      __syncthreads();
      // k4 + combine
      ode_deriv(s_ybf, s_wd1, s_wd2, s_bd1, s_bd2, s_zbf, s_k, w, lane);
      #pragma unroll
      for (int r = 0; r < 8; ++r) {
        int e = tid + r * 128; int m = e >> 6;
        float y = s_y[e] + s_dt[m] * (1.f / 6.f) * (s_acc[e] + s_k[e]);
        s_y[e] = y; s_ybf[e] = f2bf(y);
      }
      __syncthreads();
    }

    // Emit predict/loss at t = it+1.
    #pragma unroll
    for (int r = 0; r < 8; ++r) {
      int e = tid + r * 128; int m = e >> 6; int n = e & 63;
      size_t idx = ((size_t)(b0 + m) * T_ + it + 1) * D_ + n;
      float v = s_y[e];
      predict[idx] = v;
      float d = v - lfeat[idx];
      lossout[idx] = d * d * (1.f - lmask[idx]);
    }
  }
}

// ---------------------------------------------------------------------------
extern "C" void kernel_launch(void* const* d_in, const int* in_sizes, int n_in,
                              void* d_out, int out_size, void* d_ws, size_t ws_size,
                              hipStream_t stream)
{
  (void)in_sizes; (void)n_in; (void)out_size; (void)ws_size;
  const float* x     = (const float*)d_in[0];
  const int*   lens  = (const int*)  d_in[1];
  const float* lfeat = (const float*)d_in[2];
  const float* ltime = (const float*)d_in[3];
  const float* lmask = (const float*)d_in[4];
  const float* W_ih  = (const float*)d_in[5];
  const float* W_hh  = (const float*)d_in[6];
  const float* b_ih  = (const float*)d_in[7];
  const float* b_hh  = (const float*)d_in[8];
  const float* Wp1   = (const float*)d_in[9];
  const float* bp1   = (const float*)d_in[10];
  const float* Wp2   = (const float*)d_in[11];
  const float* bp2   = (const float*)d_in[12];
  const float* Wd1   = (const float*)d_in[13];
  const float* bd1   = (const float*)d_in[14];
  const float* Wd2   = (const float*)d_in[15];
  const float* bd2   = (const float*)d_in[16];

  float* predict = (float*)d_out;
  float* lossout = predict + (size_t)B_ * T_ * D_;

  // Workspace: pre-swizzled bf16 B-tiles (1.03 MB total).
  uint32_t* ws    = (uint32_t*)d_ws;
  uint32_t* wih_t = ws;                    // 64 ntiles * 5 kchunks
  uint32_t* whh_t = wih_t + 320 * 256;     // 64 * 8
  uint32_t* wp1_t = whh_t + 512 * 256;     // 16 * 8
  uint32_t* wp2_t = wp1_t + 128 * 256;     //  4 * 8
  uint32_t* wd1_t = wp2_t + 32 * 256;      // 16 * 2
  uint32_t* wd2_t = wd1_t + 32 * 256;      //  4 * 8

  swizzle_kernel<<<320, 32, 0, stream>>>(W_ih, wih_t, IN_, 5);
  swizzle_kernel<<<512, 32, 0, stream>>>(W_hh, whh_t, H_,  8);
  swizzle_kernel<<<128, 32, 0, stream>>>(Wp1,  wp1_t, H_,  8);
  swizzle_kernel<<< 32, 32, 0, stream>>>(Wp2,  wp2_t, H_,  8);
  swizzle_kernel<<< 32, 32, 0, stream>>>(Wd1,  wd1_t, D_,  2);
  swizzle_kernel<<< 32, 32, 0, stream>>>(Wd2,  wd2_t, H_,  8);

  lstm_kernel<<<B_ / 16, 256, 0, stream>>>(x, lens, lfeat, lmask, b_ih, b_hh,
                                           bp1, bp2, wih_t, whh_t, wp1_t, wp2_t,
                                           predict, lossout);
  ode_kernel<<<B_ / 16, 128, ODE_SMEM, stream>>>(ltime, lfeat, lmask, bd1, bd2,
                                                 wd1_t, wd2_t, predict, lossout);
}
